// Qwen3Attention_86474871538424
// MI455X (gfx1250) — matrix operand order repaired and checked by reading, not compiled, not run on hardware
//
#include <hip/hip_runtime.h>
#include <hip/hip_bf16.h>

#define BB 2
#define LL 2048
#define HIDN 2560
#define NH 32
#define NKV 8
#define DHD 128
#define NTOK (BB*LL)      // 4096
#define QKDIM (NH*DHD)    // 4096
#define KVDIM (NKV*DHD)   // 1024

typedef __attribute__((ext_vector_type(16))) __bf16    v16bf;
typedef __attribute__((ext_vector_type(8)))  float     v8f;
typedef __attribute__((ext_vector_type(8)))  unsigned  v8u;
typedef __attribute__((ext_vector_type(4)))  unsigned  tdm_g0_t;
typedef __attribute__((ext_vector_type(8)))  int       tdm_g1_t;
typedef __attribute__((ext_vector_type(4)))  int       tdm_g2_t;

#if defined(__has_builtin)
#if __has_builtin(__builtin_amdgcn_tensor_load_to_lds) && __has_builtin(__builtin_amdgcn_s_wait_tensorcnt)
#define HAVE_TDM 1
#else
#define HAVE_TDM 0
#endif
#else
#define HAVE_TDM 0
#endif

__device__ __forceinline__ unsigned short f2bf(float f) {
    union { float f; unsigned u; } v; v.f = f;
    unsigned r = v.u + 0x7fffu + ((v.u >> 16) & 1u);
    return (unsigned short)(r >> 16);
}
__device__ __forceinline__ unsigned ld32(const unsigned short* p) {
    return *(const unsigned*)p;
}
__device__ __forceinline__ v8f wmma_bf16(v8u a, v8u b, v8f c) {
    return __builtin_amdgcn_wmma_f32_16x16x32_bf16(
        false, __builtin_bit_cast(v16bf, a),
        false, __builtin_bit_cast(v16bf, b),
        (short)0, c, false, false);
}

#if HAVE_TDM
// Issue a TDM load: 2-byte elements, up-to-2D tile (tile_d1==0 -> 1D).
// D# group0: count=1, lds_addr, 57-bit global addr, type=2.
// D# group1: data_size=2B, tensor dims/strides, tile dims.
__device__ __forceinline__ void tdm_load(unsigned lds_addr, const void* gptr,
                                         unsigned tensor_d0, unsigned tensor_d1,
                                         unsigned tile_d0, unsigned tile_d1,
                                         unsigned stride0)
{
    unsigned long long ga = (unsigned long long)(uintptr_t)gptr;
    tdm_g0_t g0;
    g0[0] = 1u;                                   // count=1, user descriptor
    g0[1] = lds_addr;                             // LDS byte address
    g0[2] = (unsigned)(ga & 0xffffffffu);         // global addr [31:0]
    g0[3] = (unsigned)((ga >> 32) & 0x1ffffffu) | 0x80000000u; // [56:32] | type=2
    tdm_g1_t g1;
    g1[0] = 0x00010000;                           // data_size = 1 (2 bytes)
    g1[1] = (int)((tensor_d0 & 0xffffu) << 16);   // tensor_dim0[15:0] -> bits 63:48
    g1[2] = (int)((tensor_d0 >> 16) | ((tensor_d1 & 0xffffu) << 16));
    g1[3] = (int)((tensor_d1 >> 16) | (tile_d0 << 16));
    g1[4] = (int)tile_d1;                         // tile_dim1 | tile_dim2<<16
    g1[5] = (int)stride0;                         // tensor_dim0_stride[31:0]
    g1[6] = 0;
    g1[7] = 0;                                    // tensor_dim1_stride unused
    tdm_g2_t z4 = (tdm_g2_t)0;
#if __clang_major__ >= 23
    tdm_g1_t z8 = (tdm_g1_t)0;
    __builtin_amdgcn_tensor_load_to_lds(g0, g1, z4, z4, z8, 0);
#else
    __builtin_amdgcn_tensor_load_to_lds(g0, g1, z4, z4, 0);
#endif
}
#endif

// ---------------- conversion kernels ----------------
__global__ void k_cvt(const float* __restrict__ src, unsigned short* __restrict__ dst, int n) {
    for (int i = blockIdx.x * blockDim.x + threadIdx.x; i < n; i += gridDim.x * blockDim.x)
        dst[i] = f2bf(src[i]);
}
// src fp32 row-major [K][N] -> dst bf16 [N][K]
__global__ void k_cvt_t(const float* __restrict__ src, unsigned short* __restrict__ dst, int K, int N) {
    int total = K * N;
    for (int i = blockIdx.x * blockDim.x + threadIdx.x; i < total; i += gridDim.x * blockDim.x) {
        int k = i / N, n = i - k * N;
        dst[(size_t)n * K + k] = f2bf(src[i]);
    }
}

// ---------------- QKV projection: 32 tokens x 128 dims (one head) per wave ----------
// mode 0: rmsnorm + rope, out layout [b][head][l][dh]   (Q and K)
// mode 1: plain,          out layout [b][head][dh][l]   (V transposed)
__global__ __launch_bounds__(32) void k_proj(
    const unsigned short* __restrict__ xb,   // [NTOK][HIDN] bf16
    const unsigned short* __restrict__ wt,   // [nheads*128][HIDN] bf16 (transposed W)
    const float* __restrict__ normw,         // [DHD] (mode 0 only)
    const float* __restrict__ cosb,          // [NTOK][DHD]
    const float* __restrict__ sinb,          // [NTOK][DHD]
    unsigned short* __restrict__ out,
    int nheads, int mode)
{
    const int lane = threadIdx.x;
    const int ml = lane & 15, gp = lane >> 4;
    const int t0 = blockIdx.x * 32;
    const int head = blockIdx.y;

    v8f acc[2][8];
#pragma unroll
    for (int mt = 0; mt < 2; mt++)
#pragma unroll
        for (int i = 0; i < 8; i++) acc[mt][i] = (v8f)(0.0f);

    const unsigned short* arow0 = xb + (size_t)(t0 + ml) * HIDN;
    const unsigned short* arow1 = xb + (size_t)(t0 + 16 + ml) * HIDN;
    const int ncol0 = head * DHD + ml;

    for (int kk = 0; kk < HIDN; kk += 32) {
        v8u au0, au1;
#pragma unroll
        for (int r = 0; r < 8; r++) {
            int k = kk + ((r >> 2) << 4) + (gp << 3) + ((r & 3) << 1);
            au0[r] = ld32(arow0 + k);
            au1[r] = ld32(arow1 + k);
        }
#pragma unroll
        for (int nt = 0; nt < 8; nt++) {
            const unsigned short* brow = wt + (size_t)(ncol0 + nt * 16) * HIDN;
            v8u bu;
#pragma unroll
            for (int r = 0; r < 8; r++) {
                int k = kk + ((r >> 2) << 4) + (gp << 3) + ((r & 3) << 1);
                bu[r] = ld32(brow + k);
            }
            acc[0][nt] = wmma_bf16(au0, bu, acc[0][nt]);
            acc[1][nt] = wmma_bf16(au1, bu, acc[1][nt]);
        }
    }

    __shared__ float sc[32 * 132];
    if (mode == 0) {
#pragma unroll
        for (int mt = 0; mt < 2; mt++)
#pragma unroll
            for (int nt = 0; nt < 8; nt++)
#pragma unroll
                for (int r = 0; r < 8; r++)
                    sc[(mt * 16 + gp * 8 + r) * 132 + nt * 16 + ml] = acc[mt][nt][r];
        __syncthreads();

        // each lane owns one full token row of 128 dims
        const int row = lane;
        const int tok = t0 + row;
        float* base = sc + row * 132;
        float ss = 0.f;
        for (int i = 0; i < 128; i++) { float v = base[i]; ss += v * v; }
        float rs = __frsqrt_rn(ss * (1.0f / 128.0f) + 1e-6f);

        const int b = tok >> 11, l = tok & (LL - 1);
        const float* crow = cosb + (size_t)tok * DHD;
        const float* srow = sinb + (size_t)tok * DHD;
        unsigned short* orow = out + (((size_t)(b * nheads + head)) * LL + l) * DHD;
        for (int d = 0; d < DHD; d += 2) {
            float v0 = base[d]     * rs * normw[d];
            float v1 = base[d + 1] * rs * normw[d + 1];
            int pd0 = (d + 64) & 127, pd1 = (d + 65) & 127;
            float p0 = base[pd0] * rs * normw[pd0];
            float p1 = base[pd1] * rs * normw[pd1];
            float r0 = (d < 64) ? -p0 : p0;
            float r1 = (d + 1 < 64) ? -p1 : p1;
            unsigned pack = (unsigned)f2bf(v0 * crow[d]     + r0 * srow[d])
                          | ((unsigned)f2bf(v1 * crow[d + 1] + r1 * srow[d + 1]) << 16);
            *(unsigned*)(orow + d) = pack;
        }
    } else {
        const int b = t0 >> 11, lbase = t0 & (LL - 1);
#pragma unroll
        for (int nt = 0; nt < 8; nt++) {
            int dh = nt * 16 + ml;
            unsigned short* ocol = out + (((size_t)(b * nheads + head)) * DHD + dh) * LL + lbase;
#pragma unroll
            for (int mt = 0; mt < 2; mt++)
#pragma unroll
                for (int r = 0; r < 8; r++)
                    ocol[mt * 16 + gp * 8 + r] = f2bf(acc[mt][nt][r]);
        }
    }
}

// ---------------- causal GQA flash attention ----------------
// One block = (b, kv-head, 16-query tile); 4 waves = the 4 grouped Q heads.
// K/V panels double-buffered in LDS (TDM prefetch pipelined with WMMA compute).
__global__ __launch_bounds__(128) void k_attn(
    const unsigned short* __restrict__ qb,   // [b][H][L][DH]
    const unsigned short* __restrict__ kbuf, // [b][HKV][L][DH]
    const unsigned short* __restrict__ vtb,  // [b][HKV][DH][L]
    unsigned short* __restrict__ attn)       // [tok][H*DH]
{
    const int tid  = threadIdx.x;
    const int wid  = tid >> 5;
    const int lane = tid & 31;
    const int ml = lane & 15, gp = lane >> 4;
    const int qt  = blockIdx.x;   // 16-token query tile
    const int kvh = blockIdx.y;
    const int b   = blockIdx.z;
    const int h   = kvh * 4 + wid; // G = 4

    const unsigned short* qptr = qb   + ((size_t)(b * NH  + h  ) * LL) * DHD;
    const unsigned short* kptr = kbuf + ((size_t)(b * NKV + kvh) * LL) * DHD;
    const unsigned short* vptr = vtb  + ((size_t)(b * NKV + kvh) * DHD) * LL;

    __shared__ unsigned short kp[2][32 * DHD];  // [buf][key][dh] 2 x 8 KB
    __shared__ unsigned short vp[2][DHD * 32];  // [buf][dh][key] 2 x 8 KB
    __shared__ unsigned short plds[4][16 * 32]; // per-wave P transpose (wave-private)

    // stage K panel (contiguous 32x128) and V panel (2D 128x32) into LDS buffer bi
    auto stage = [&](int bi, int key0s) {
#if HAVE_TDM
        if (tid < 32) {
            tdm_load((unsigned)(uintptr_t)(void*)&kp[bi][0], kptr + (size_t)key0s * DHD,
                     32 * DHD, 1, 32 * DHD, 0, 32 * DHD);        // 1D copy
            tdm_load((unsigned)(uintptr_t)(void*)&vp[bi][0], vptr + key0s,
                     LL, DHD, 32, DHD, LL);                      // 2D: 128 rows x 32
        }
#else
        const uint4* gk = (const uint4*)(kptr + (size_t)key0s * DHD);
        uint4* lk = (uint4*)&kp[bi][0];
#pragma unroll
        for (int i = 0; i < 4; i++) lk[tid + i * 128] = gk[tid + i * 128];
        const uint4* gv = (const uint4*)(vptr + (size_t)tid * LL + key0s);
        uint4* lv = (uint4*)(&vp[bi][0] + tid * 32);
#pragma unroll
        for (int i = 0; i < 4; i++) lv[i] = gv[i];
#endif
    };

    v8u qa[4];
#pragma unroll
    for (int ks = 0; ks < 4; ks++)
#pragma unroll
        for (int r = 0; r < 8; r++) {
            int k = ks * 32 + ((r >> 2) << 4) + (gp << 3) + ((r & 3) << 1);
            qa[ks][r] = ld32(qptr + (size_t)(qt * 16 + ml) * DHD + k);
        }

    v8f oacc[8];
#pragma unroll
    for (int i = 0; i < 8; i++) oacc[i] = (v8f)(0.0f);
    float mrow[8], lrow[8];
#pragma unroll
    for (int r = 0; r < 8; r++) { mrow[r] = -1e30f; lrow[r] = 0.f; }

    const float scale = 0.08838834764831845f; // 1/sqrt(128)
    const int nblk = (qt >> 1) + 1;           // 32-key blocks, causal

    stage(0, 0);                              // prefetch first block
    for (int ib = 0; ib < nblk; ib++) {
        const int cur  = ib & 1;
        const int key0 = ib * 32;

#if HAVE_TDM
        if (tid < 32) __builtin_amdgcn_s_wait_tensorcnt((short)0);
#endif
        __syncthreads();                      // buf[cur] ready; all readers of it done
        if (ib + 1 < nblk) stage(1 - cur, key0 + 32);  // prefetch next, overlap compute

        const unsigned short* kpc = &kp[cur][0];
        const unsigned short* vpc = &vp[cur][0];

        // ---- scores: Q @ K^T for keys [key0, key0+32)
        v8f s0 = (v8f)(0.0f), s1 = (v8f)(0.0f);
#pragma unroll
        for (int ks = 0; ks < 4; ks++) {
            v8u b0, b1;
#pragma unroll
            for (int r = 0; r < 8; r++) {
                int kd = ks * 32 + ((r >> 2) << 4) + (gp << 3) + ((r & 3) << 1);
                b0[r] = *(const unsigned*)(kpc + (ml)      * DHD + kd);
                b1[r] = *(const unsigned*)(kpc + (ml + 16) * DHD + kd);
            }
            s0 = wmma_bf16(qa[ks], b0, s0);
            s1 = wmma_bf16(qa[ks], b1, s1);
        }

        float p0[8], p1[8], alpha[8];
#pragma unroll
        for (int r = 0; r < 8; r++) {
            int rowtok = qt * 16 + gp * 8 + r;
            p0[r] = (key0      + ml <= rowtok) ? s0[r] * scale : -1e30f;
            p1[r] = (key0 + 16 + ml <= rowtok) ? s1[r] * scale : -1e30f;
        }
#pragma unroll
        for (int r = 0; r < 8; r++) {
            float mx = fmaxf(p0[r], p1[r]);
            mx = fmaxf(mx, __shfl_xor(mx, 1, 32));
            mx = fmaxf(mx, __shfl_xor(mx, 2, 32));
            mx = fmaxf(mx, __shfl_xor(mx, 4, 32));
            mx = fmaxf(mx, __shfl_xor(mx, 8, 32));
            float mnew = fmaxf(mrow[r], mx);
            alpha[r] = __expf(mrow[r] - mnew);
            p0[r] = __expf(p0[r] - mnew);
            p1[r] = __expf(p1[r] - mnew);
            float rsum = p0[r] + p1[r];
            rsum += __shfl_xor(rsum, 1, 32);
            rsum += __shfl_xor(rsum, 2, 32);
            rsum += __shfl_xor(rsum, 4, 32);
            rsum += __shfl_xor(rsum, 8, 32);
            lrow[r] = lrow[r] * alpha[r] + rsum;
            mrow[r] = mnew;
        }
#pragma unroll
        for (int nt = 0; nt < 8; nt++)
#pragma unroll
            for (int r = 0; r < 8; r++)
                oacc[nt][r] *= alpha[r];

        // ---- transpose P (C-layout) into A-fragment layout via wave-private LDS.
        // Same-wave LDS ops execute in order; only a compiler fence is needed.
        unsigned short* pl = plds[wid];
#pragma unroll
        for (int r = 0; r < 8; r++) {
            pl[(gp * 8 + r) * 32      + ml] = f2bf(p0[r]);
            pl[(gp * 8 + r) * 32 + 16 + ml] = f2bf(p1[r]);
        }
        __builtin_amdgcn_wave_barrier();
        v8u pu;
#pragma unroll
        for (int r = 0; r < 8; r++) {
            int kc = ((r >> 2) << 4) + (gp << 3) + ((r & 3) << 1);
            pu[r] = *(const unsigned*)(pl + ml * 32 + kc);
        }

        // ---- O += P @ V
#pragma unroll
        for (int nt = 0; nt < 8; nt++) {
            v8u vu;
#pragma unroll
            for (int r = 0; r < 8; r++) {
                int dh = nt * 16 + ml;
                int kl = ((r >> 2) << 4) + (gp << 3) + ((r & 3) << 1);
                vu[r] = *(const unsigned*)(vpc + dh * 32 + kl);
            }
            oacc[nt] = wmma_bf16(pu, vu, oacc[nt]);
        }
        // no bottom barrier: next iteration's top barrier separates these reads
        // from the TDM issue that overwrites buf[cur]
    }

#pragma unroll
    for (int r = 0; r < 8; r++) lrow[r] = 1.0f / lrow[r];
#pragma unroll
    for (int nt = 0; nt < 8; nt++)
#pragma unroll
        for (int r = 0; r < 8; r++) {
            int tok = b * LL + qt * 16 + gp * 8 + r;
            int col = h * DHD + nt * 16 + ml;
            attn[(size_t)tok * QKDIM + col] = f2bf(oacc[nt][r] * lrow[r]);
        }
}

// ---------------- output projection: attn(bf16) @ Wo -> fp32, 32 tokens/wave -------
__global__ __launch_bounds__(32) void k_ogemm(
    const unsigned short* __restrict__ ab,   // [NTOK][QKDIM]
    const unsigned short* __restrict__ wot,  // [HIDN][QKDIM] (transposed Wo)
    float* __restrict__ out)                 // [NTOK][HIDN]
{
    const int lane = threadIdx.x;
    const int ml = lane & 15, gp = lane >> 4;
    const int t0 = blockIdx.x * 32;
    const int c0 = blockIdx.y * 128;

    v8f acc[2][8];
#pragma unroll
    for (int mt = 0; mt < 2; mt++)
#pragma unroll
        for (int i = 0; i < 8; i++) acc[mt][i] = (v8f)(0.0f);

    const unsigned short* arow0 = ab + (size_t)(t0 + ml) * QKDIM;
    const unsigned short* arow1 = ab + (size_t)(t0 + 16 + ml) * QKDIM;
    for (int kk = 0; kk < QKDIM; kk += 32) {
        v8u au0, au1;
#pragma unroll
        for (int r = 0; r < 8; r++) {
            int k = kk + ((r >> 2) << 4) + (gp << 3) + ((r & 3) << 1);
            au0[r] = ld32(arow0 + k);
            au1[r] = ld32(arow1 + k);
        }
#pragma unroll
        for (int nt = 0; nt < 8; nt++) {
            const unsigned short* brow = wot + (size_t)(c0 + nt * 16 + ml) * QKDIM;
            v8u bu;
#pragma unroll
            for (int r = 0; r < 8; r++) {
                int k = kk + ((r >> 2) << 4) + (gp << 3) + ((r & 3) << 1);
                bu[r] = ld32(brow + k);
            }
            acc[0][nt] = wmma_bf16(au0, bu, acc[0][nt]);
            acc[1][nt] = wmma_bf16(au1, bu, acc[1][nt]);
        }
    }
#pragma unroll
    for (int mt = 0; mt < 2; mt++)
#pragma unroll
        for (int nt = 0; nt < 8; nt++)
#pragma unroll
            for (int r = 0; r < 8; r++)
                out[(size_t)(t0 + mt * 16 + gp * 8 + r) * HIDN + c0 + nt * 16 + ml]
                    = acc[mt][nt][r];
}

// ---------------- host launcher ----------------
extern "C" void kernel_launch(void* const* d_in, const int* in_sizes, int n_in,
                              void* d_out, int out_size, void* d_ws, size_t ws_size,
                              hipStream_t stream) {
    (void)in_sizes; (void)n_in; (void)out_size; (void)ws_size;
    const float* x    = (const float*)d_in[0];
    const float* Wq   = (const float*)d_in[1];
    const float* Wk   = (const float*)d_in[2];
    const float* Wv   = (const float*)d_in[3];
    const float* Wo   = (const float*)d_in[4];
    const float* qnw  = (const float*)d_in[5];
    const float* knw  = (const float*)d_in[6];
    const float* cosb = (const float*)d_in[7];
    const float* sinb = (const float*)d_in[8];

    char* ws = (char*)d_ws;
    size_t off = 0;
    auto alloc = [&](size_t elems) {
        unsigned short* p = (unsigned short*)(ws + off);
        off += elems * 2;
        return p;
    };
    unsigned short* xb  = alloc((size_t)NTOK * HIDN);         // x bf16
    unsigned short* wqt = alloc((size_t)QKDIM * HIDN);        // Wq^T bf16
    unsigned short* wkt = alloc((size_t)KVDIM * HIDN);        // Wk^T bf16
    unsigned short* wvt = alloc((size_t)KVDIM * HIDN);        // Wv^T bf16
    unsigned short* wot = alloc((size_t)HIDN * QKDIM);        // Wo^T bf16
    unsigned short* qbf = alloc((size_t)BB * NH  * LL * DHD); // q post norm+rope
    unsigned short* kbf = alloc((size_t)BB * NKV * LL * DHD); // k post norm+rope
    unsigned short* vtb = alloc((size_t)BB * NKV * DHD * LL); // v transposed
    unsigned short* att = alloc((size_t)NTOK * QKDIM);        // attention output

    k_cvt  <<<4096, 256, 0, stream>>>(x,  xb,  NTOK * HIDN);
    k_cvt_t<<<4096, 256, 0, stream>>>(Wq, wqt, HIDN, QKDIM);
    k_cvt_t<<<2048, 256, 0, stream>>>(Wk, wkt, HIDN, KVDIM);
    k_cvt_t<<<2048, 256, 0, stream>>>(Wv, wvt, HIDN, KVDIM);
    k_cvt_t<<<4096, 256, 0, stream>>>(Wo, wot, QKDIM, HIDN);

    k_proj<<<dim3(NTOK / 32, NH ), 32, 0, stream>>>(xb, wqt, qnw, cosb, sinb, qbf, NH, 0);
    k_proj<<<dim3(NTOK / 32, NKV), 32, 0, stream>>>(xb, wkt, knw, cosb, sinb, kbf, NKV, 0);
    k_proj<<<dim3(NTOK / 32, NKV), 32, 0, stream>>>(xb, wvt, nullptr, cosb, sinb, vtb, NKV, 1);

    k_attn<<<dim3(LL / 16, NKV, BB), 128, 0, stream>>>(qbf, kbf, vtb, att);

    k_ogemm<<<dim3(NTOK / 32, HIDN / 128), 32, 0, stream>>>(att, wot, (float*)d_out);
}
